// VQVAE_5317169513003
// MI455X (gfx1250) — matrix operand order repaired and checked by reading, not compiled, not run on hardware
//
#include <hip/hip_runtime.h>
#include <hip/hip_bf16.h>

// ---------------------------------------------------------------------------
// VQ-VAE forward for MI455X (gfx1250, wave32).
// Heavy GEMMs (convs, transposed conv, codebook scores) run as implicit-GEMM
// v_wmma_f32_16x16x32_bf16 (bf16 in, fp32 accumulate), 4 N-accumulators per
// wave reusing one A fragment.  All spatial/channel dims are powers of two,
// so every index is compile-time shift/mask (no integer division, 32-bit
// offsets only).  Boundary 1-channel convs + VQ argmin/loss are fp32 VALU.
// ---------------------------------------------------------------------------

typedef __attribute__((ext_vector_type(16))) __bf16 bf16x16;
typedef __attribute__((ext_vector_type(8)))  float  f32x8;

union FragU { unsigned u[8]; bf16x16 v; };

__device__ __forceinline__ unsigned short f2bf(float f) {
  unsigned u = __float_as_uint(f);
  u += 0x7FFFu + ((u >> 16) & 1u);          // round-to-nearest-even
  return (unsigned short)(u >> 16);
}

// ---------------------------------------------------------------------------
// Weight repack: OIHW fp32 -> row-major bf16 [M x Kpad], K = Cin*KH*KW.
// transp_flip=1 handles torch ConvTranspose2d layout (I,O,kh,kw) with kernel
// flip:  A[m][ci,ky,kx] = w[ci][m][KH-1-ky][KW-1-kx].   (tiny, one-shot)
// ---------------------------------------------------------------------------
__global__ void repack_w(const float* __restrict__ w, unsigned short* __restrict__ wp,
                         int M, int Cin, int KH, int KW, int transp_flip)
{
  const int Kpad = Cin * KH * KW;
  const int total = M * Kpad;
  int idx = blockIdx.x * 256 + threadIdx.x;
  if (idx >= total) return;
  int m = idx / Kpad, k = idx - m * Kpad;
  int khkw = KH * KW;
  int ci = k / khkw, rr = k - ci * khkw, ky = rr / KW, kx = rr - ky * KW;
  float v;
  if (!transp_flip) v = w[(((size_t)m * Cin + ci) * KH + ky) * KW + kx];
  else              v = w[(((size_t)ci * M + m) * KH + (KH - 1 - ky)) * KW + (KW - 1 - kx)];
  wp[idx] = f2bf(v);
}

// ---------------------------------------------------------------------------
// Activation gather: compile-time dilation/relu/spatial-shift.
// LDIL==2: gather from zero-dilated input (transposed conv, parity mask).
// ---------------------------------------------------------------------------
template <int LDIL, int PRERELU, int IWL2, int IHL2>
__device__ __forceinline__ float fetchv(const float* __restrict__ inb, unsigned cioff,
                                        int ty, int tx)
{
  bool ok;
  if (LDIL == 2) {
    int pm = ty | tx;
    ok = (pm >= 0) && !(pm & 1);
    ty >>= 1; tx >>= 1;
    ok = ok && (ty < (1 << IHL2)) && (tx < (1 << IWL2));
  } else {
    ok = ((unsigned)ty < (1u << IHL2)) && ((unsigned)tx < (1u << IWL2));
  }
  float v = 0.f;
  if (ok) {
    v = inb[cioff + ((unsigned)ty << IWL2) + (unsigned)tx];
    if (PRERELU) v = fmaxf(v, 0.f);
  }
  return v;
}

// ---------------------------------------------------------------------------
// Implicit-GEMM conv via WMMA.  All divisors/dims are template constants:
// k->(ci,ky,kx) decode and n->(b,oy,ox) decode are shift/mask, addresses are
// 32-bit offsets.  Block = 128 threads = 4 waves; each wave computes a
// 16M x 64N tile (4 accumulators sharing one A fragment).
// grid.x = N/256, grid.y = Cout/16.  N is a multiple of 256 for every layer,
// so EXEC stays all-ones through the WMMA region (ISA requirement).
// store_mode==0: NCHW output + optional bias / residual-add.
// store_mode==1: transposed score matrix out[n*Mpad + m] (VQ).
// ---------------------------------------------------------------------------
template <int KHKW, int KW, int LDIL, int PRERELU, int IWL2, int IHL2, int OWL2, int OHL2>
__global__ void __launch_bounds__(128) conv_wmma_bf16(
    const float* __restrict__ in, const unsigned short* __restrict__ wpack,
    const float* __restrict__ bias, const float* __restrict__ resid,
    float* __restrict__ out,
    int Cin, int Cout, int stride, int pad, int Kpad, int store_mode, int Mpad)
{
  constexpr int CSL2  = IWL2 + IHL2;   // log2(channel stride)
  constexpr int OHWL2 = OWL2 + OHL2;   // log2(OH*OW)

  const int lane  = threadIdx.x & 31;
  const int wave  = threadIdx.x >> 5;
  const int lrow  = lane & 15;
  const int lhalf = lane >> 4;
  const int nbase = blockIdx.x * 256 + wave * 64;

  // per-subtile B-column coordinates for this lane (pure shift/mask)
  unsigned base[4]; int bb[4], tt[4], oys[4], oxs[4];
#pragma unroll
  for (int s = 0; s < 4; ++s) {
    int n  = nbase + s * 16 + lrow;
    int b_ = n >> OHWL2;
    int t_ = n & ((1 << OHWL2) - 1);
    int oy = t_ >> OWL2;
    int ox = t_ & ((1 << OWL2) - 1);
    bb[s] = b_; tt[s] = t_;
    base[s] = (unsigned)(b_ * Cin) << CSL2;
    oys[s] = oy * stride - pad;
    oxs[s] = ox * stride - pad;
  }

  const int mrow = blockIdx.y * 16 + lrow;
  // A-fragment layout (16-bit A 16x32): this lane's 8 bf16 pairs are two
  // contiguous 16B runs at (k0 + lhalf*8) and (k0 + 16 + lhalf*8).
  const unsigned short* wrowp = wpack + (size_t)mrow * Kpad + lhalf * 8;

  f32x8 acc[4];
#pragma unroll
  for (int s = 0; s < 4; ++s) acc[s] = (f32x8){0.f,0.f,0.f,0.f,0.f,0.f,0.f,0.f};

  for (int k0 = 0; k0 < Kpad; k0 += 32) {
    FragU a;
    const uint4 alo = *(const uint4*)(wrowp + k0);
    const uint4 ahi = *(const uint4*)(wrowp + k0 + 16);
    a.u[0] = alo.x; a.u[1] = alo.y; a.u[2] = alo.z; a.u[3] = alo.w;
    a.u[4] = ahi.x; a.u[5] = ahi.y; a.u[6] = ahi.z; a.u[7] = ahi.w;

    // decode this lane's 16 B-row K indices once (constant divisors)
    unsigned co0[8], co1[8]; int dy0[8], dx0[8], dy1[8], dx1[8];
#pragma unroll
    for (int j = 0; j < 8; ++j) {
      // B K order matches A: j<4 -> K=lhalf*8+2j, j>=4 -> K=16+lhalf*8+2(j-4)
      const int kk = k0 + lhalf * 8 + 2 * j + ((j >= 4) ? 8 : 0);
      {
        int ci = kk / KHKW, r = kk - ci * KHKW;
        dy0[j] = r / KW; dx0[j] = r - dy0[j] * KW;
        co0[j] = (unsigned)ci << CSL2;
      }
      {
        int k1 = kk + 1;
        int ci = k1 / KHKW, r = k1 - ci * KHKW;
        dy1[j] = r / KW; dx1[j] = r - dy1[j] * KW;
        co1[j] = (unsigned)ci << CSL2;
      }
    }

#pragma unroll
    for (int s = 0; s < 4; ++s) {
      const float* __restrict__ inb = in + base[s];
      FragU b;
#pragma unroll
      for (int j = 0; j < 8; ++j) {
        float v0 = fetchv<LDIL, PRERELU, IWL2, IHL2>(inb, co0[j], oys[s] + dy0[j], oxs[s] + dx0[j]);
        float v1 = fetchv<LDIL, PRERELU, IWL2, IHL2>(inb, co1[j], oys[s] + dy1[j], oxs[s] + dx1[j]);
        b.u[j] = (unsigned)f2bf(v0) | ((unsigned)f2bf(v1) << 16);
      }
      acc[s] = __builtin_amdgcn_wmma_f32_16x16x32_bf16(false, a.v, false, b.v,
                                                       (short)0, acc[s], false, false);
    }
  }

  // C/D layout: VGPR r -> M = r + 8*lhalf, N = lrow
#pragma unroll
  for (int s = 0; s < 4; ++s) {
    const unsigned n = (unsigned)(nbase + s * 16 + lrow);
#pragma unroll
    for (int r = 0; r < 8; ++r) {
      const int m = blockIdx.y * 16 + lhalf * 8 + r;
      float val = acc[s][r];
      if (store_mode == 1) {
        out[n * (unsigned)Mpad + (unsigned)m] = val;
      } else {
        if (bias)  val += bias[m];
        const unsigned oidx = ((unsigned)(bb[s] * Cout + m) << OHWL2) + (unsigned)tt[s];
        if (resid) val += resid[oidx];
        out[oidx] = val;
      }
    }
  }
}

// ---------------------------------------------------------------------------
// enc_c1: 1->64 ch, 4x4 s2 p1 + bias + relu.  K=16: pure bandwidth, fp32 VALU.
// x:(16,1,256,256)  out:(16,64,128,128)
// ---------------------------------------------------------------------------
__global__ void enc_c1_direct(const float* __restrict__ x, const float* __restrict__ w,
                              const float* __restrict__ b, float* __restrict__ out)
{
  int idx = blockIdx.x * 256 + threadIdx.x;
  if (idx >= 16 * 64 * 128 * 128) return;
  int ox = idx & 127; int t = idx >> 7;
  int oy = t & 127;   t >>= 7;
  int co = t & 63;    int bb = t >> 6;
  const float* xb = x + (size_t)bb * 65536;
  float s = b[co];
#pragma unroll
  for (int ky = 0; ky < 4; ++ky) {
    int iy = oy * 2 - 1 + ky;
    if ((unsigned)iy >= 256u) continue;
#pragma unroll
    for (int kx = 0; kx < 4; ++kx) {
      int ix = ox * 2 - 1 + kx;
      if ((unsigned)ix >= 256u) continue;
      s += xb[iy * 256 + ix] * w[co * 16 + ky * 4 + kx];
    }
  }
  out[idx] = fmaxf(s, 0.f);
}

// ---------------------------------------------------------------------------
// dec_t2: transposed conv 64->1, 4x4 s2 p1 + bias.  M=1: fp32 VALU.
// in:(16,64,128,128)  w:(64,1,4,4) torch layout  out:(16,1,256,256)
// ---------------------------------------------------------------------------
__global__ void dec_t2_direct(const float* __restrict__ in, const float* __restrict__ w,
                              const float* __restrict__ bias, float* __restrict__ out)
{
  int idx = blockIdx.x * 256 + threadIdx.x;
  if (idx >= 16 * 256 * 256) return;
  int ox = idx & 255; int t = idx >> 8;
  int oy = t & 255;   int bb = t >> 8;
  float s = bias[0];
  for (int ky = 0; ky < 4; ++ky) {
    int ty = oy + ky - 2;
    if (ty < 0 || (ty & 1)) continue;
    int iy = ty >> 1;
    if (iy >= 128) continue;
    for (int kx = 0; kx < 4; ++kx) {
      int tx = ox + kx - 2;
      if (tx < 0 || (tx & 1)) continue;
      int ix = tx >> 1;
      if (ix >= 128) continue;
      for (int ci = 0; ci < 64; ++ci) {
        s += in[(((unsigned)bb * 64 + ci) << 14) + (iy << 7) + ix]
           * w[ci * 16 + (3 - ky) * 4 + (3 - kx)];
      }
    }
  }
  out[idx] = s;
}

// ---------------------------------------------------------------------------
// VQ support kernels (fp32)
// ---------------------------------------------------------------------------
__global__ void cb_norms(const float* __restrict__ cb, float* __restrict__ cn)
{
  int m = blockIdx.x * 256 + threadIdx.x;
  if (m >= 512) return;
  float s = 0.f;
  for (int e = 0; e < 64; ++e) { float v = cb[m * 64 + e]; s += v * v; }
  cn[m] = s;
}

// dist(n,m) = ||z||^2 + ||c_m||^2 - 2 S[n][m];  argmin over m == argmin(cn - 2S)
__global__ void vq_argmin(const float* __restrict__ S, const float* __restrict__ cn,
                          int* __restrict__ idx)
{
  int n = blockIdx.x * 256 + threadIdx.x;
  if (n >= 65536) return;
  const float* Sn = S + (size_t)n * 512;
  float best = cn[0] - 2.f * Sn[0];
  int bi = 0;
  for (int m = 1; m < 512; ++m) {
    float d = cn[m] - 2.f * Sn[m];
    if (d < best) { best = d; bi = m; }   // first-min tie-break (jnp.argmin)
  }
  idx[n] = bi;
}

// qz = codebook[idx] (straight-through forward value); loss partial sums.
// z,qz: (16,64,64,64) NCHW; n = b*4096 + y*64 + x
__global__ void vq_gather_loss(const int* __restrict__ idx, const float* __restrict__ cb,
                               const float* __restrict__ z, float* __restrict__ qz,
                               float* __restrict__ lacc)
{
  int n = blockIdx.x * 256 + threadIdx.x;
  if (n >= 65536) return;
  int bb = n >> 12, t = n & 4095;
  const float* c = cb + (size_t)idx[n] * 64;
  float s = 0.f;
  for (int e = 0; e < 64; ++e) {
    unsigned zi = (((unsigned)bb * 64 + e) << 12) + t;
    float ce = c[e];
    float d = ce - z[zi];
    s += d * d;
    qz[zi] = ce;
  }
  atomicAdd(lacc, s);
}

// loss = q_latent + 0.25*e_latent = 1.25 * mean((q - ze)^2), mean over 65536*64
__global__ void vq_finalize(const float* __restrict__ lacc, float* __restrict__ out)
{
  out[0] = 1.25f * lacc[0] / 4194304.f;
}

// ---------------------------------------------------------------------------
// Host-side orchestration
// ---------------------------------------------------------------------------
#define CONV(KHKW_, KW_, LDIL_, PRERELU_, IWL2_, IHL2_, OWL2_, OHL2_,               \
             in_, wp_, bias_, resid_, out_, Cin_, Cout_, stride_, pad_, sm_)        \
  do {                                                                              \
    long N_ = 16L << ((OWL2_) + (OHL2_));                                           \
    dim3 g_((unsigned)(N_ / 256), (unsigned)((Cout_) / 16));                        \
    conv_wmma_bf16<KHKW_, KW_, LDIL_, PRERELU_, IWL2_, IHL2_, OWL2_, OHL2_>         \
        <<<g_, 128, 0, stream>>>(in_, wp_, bias_, resid_, out_,                     \
                                 Cin_, Cout_, stride_, pad_,                        \
                                 (Cin_) * (KHKW_), sm_, Cout_);                     \
  } while (0)

extern "C" void kernel_launch(void* const* d_in, const int* in_sizes, int n_in,
                              void* d_out, int out_size, void* d_ws, size_t ws_size,
                              hipStream_t stream)
{
  (void)in_sizes; (void)n_in; (void)out_size; (void)ws_size;
  const float* x      = (const float*)d_in[0];
  const float* ec1_w  = (const float*)d_in[1];
  const float* ec1_b  = (const float*)d_in[2];
  const float* ec2_w  = (const float*)d_in[3];
  const float* ec2_b  = (const float*)d_in[4];
  const float* er1_w1 = (const float*)d_in[5];
  const float* er1_w2 = (const float*)d_in[6];
  const float* er2_w1 = (const float*)d_in[7];
  const float* er2_w2 = (const float*)d_in[8];
  const float* pq_w   = (const float*)d_in[9];
  const float* pq_b   = (const float*)d_in[10];
  const float* cb     = (const float*)d_in[11];
  const float* dc1_w  = (const float*)d_in[12];
  const float* dc1_b  = (const float*)d_in[13];
  const float* dr1_w1 = (const float*)d_in[14];
  const float* dr1_w2 = (const float*)d_in[15];
  const float* dr2_w1 = (const float*)d_in[16];
  const float* dr2_w2 = (const float*)d_in[17];
  const float* dt1_w  = (const float*)d_in[18];
  const float* dt1_b  = (const float*)d_in[19];
  const float* dt2_w  = (const float*)d_in[20];
  const float* dt2_b  = (const float*)d_in[21];
  float* y_out = (float*)d_out;

  // ---- workspace layout (fp32 units) ----
  float* ws = (float*)d_ws;
  size_t o = 0;
  float* t_a1 = ws + o; o += (size_t)16 * 64 * 128 * 128;   // 16.7M
  float* t_p  = ws + o; o += (size_t)16 * 128 * 64 * 64;    // 8.4M
  float* t_q  = ws + o; o += (size_t)16 * 128 * 64 * 64;
  float* t_r  = ws + o; o += (size_t)16 * 32 * 64 * 64;
  float* t_z  = ws + o; o += (size_t)16 * 64 * 64 * 64;
  float* t_qz = ws + o; o += (size_t)16 * 64 * 64 * 64;
  float* t_S  = ws + o; o += (size_t)65536 * 512;           // VQ scores [N x 512]
  float* t_cn = ws + o; o += 512;
  int*   t_ix = (int*)(ws + o); o += 65536;
  float* t_ls = ws + o; o += 16;
  unsigned short* pk = (unsigned short*)(ws + o);
  size_t po = 0;
  unsigned short* p_ec2  = pk + po; po += (size_t)128 * 1024;
  unsigned short* p_r1a  = pk + po; po += (size_t)32 * 1152;
  unsigned short* p_r1b  = pk + po; po += (size_t)128 * 32;
  unsigned short* p_r2a  = pk + po; po += (size_t)32 * 1152;
  unsigned short* p_r2b  = pk + po; po += (size_t)128 * 32;
  unsigned short* p_pq   = pk + po; po += (size_t)64 * 128;
  unsigned short* p_cb   = pk + po; po += (size_t)512 * 64;
  unsigned short* p_dc1  = pk + po; po += (size_t)128 * 576;
  unsigned short* p_dr1a = pk + po; po += (size_t)32 * 1152;
  unsigned short* p_dr1b = pk + po; po += (size_t)128 * 32;
  unsigned short* p_dr2a = pk + po; po += (size_t)32 * 1152;
  unsigned short* p_dr2b = pk + po; po += (size_t)128 * 32;
  unsigned short* p_dt1  = pk + po; po += (size_t)64 * 2048;

  auto repack = [&](const float* w, unsigned short* dst, int M, int Cin, int KH, int KW, int tf) {
    int total = M * Cin * KH * KW;
    repack_w<<<(total + 255) / 256, 256, 0, stream>>>(w, dst, M, Cin, KH, KW, tf);
  };

  // ---- weight repacks (bf16) ----
  repack(ec2_w,  p_ec2, 128,  64, 4, 4, 0);
  repack(er1_w1, p_r1a,  32, 128, 3, 3, 0);
  repack(er1_w2, p_r1b, 128,  32, 1, 1, 0);
  repack(er2_w1, p_r2a,  32, 128, 3, 3, 0);
  repack(er2_w2, p_r2b, 128,  32, 1, 1, 0);
  repack(pq_w,   p_pq,   64, 128, 1, 1, 0);
  repack(cb,     p_cb,  512,  64, 1, 1, 0);
  repack(dc1_w,  p_dc1, 128,  64, 3, 3, 0);
  repack(dr1_w1, p_dr1a, 32, 128, 3, 3, 0);
  repack(dr1_w2, p_dr1b,128,  32, 1, 1, 0);
  repack(dr2_w1, p_dr2a, 32, 128, 3, 3, 0);
  repack(dr2_w2, p_dr2b,128,  32, 1, 1, 0);
  repack(dt1_w,  p_dt1,  64, 128, 4, 4, 1);   // transposed + flipped

  // ---- encoder ----
  enc_c1_direct<<<65536, 256, 0, stream>>>(x, ec1_w, ec1_b, t_a1);
  //          khkw kw ld pr iw ih ow oh
  CONV(16, 4, 1, 0, 7, 7, 6, 6, t_a1, p_ec2, ec2_b, nullptr, t_p, 64, 128, 2, 1, 0);
  CONV( 9, 3, 1, 1, 6, 6, 6, 6, t_p,  p_r1a, nullptr, nullptr, t_r, 128, 32, 1, 1, 0);
  CONV( 1, 1, 1, 1, 6, 6, 6, 6, t_r,  p_r1b, nullptr, t_p,     t_q,  32, 128, 1, 0, 0);
  CONV( 9, 3, 1, 1, 6, 6, 6, 6, t_q,  p_r2a, nullptr, nullptr, t_r, 128, 32, 1, 1, 0);
  CONV( 1, 1, 1, 1, 6, 6, 6, 6, t_r,  p_r2b, nullptr, t_q,     t_p,  32, 128, 1, 0, 0);
  CONV( 1, 1, 1, 0, 6, 6, 6, 6, t_p,  p_pq,  pq_b,   nullptr,  t_z, 128,  64, 1, 0, 0);

  // ---- vector quantizer ----
  CONV( 1, 1, 1, 0, 6, 6, 6, 6, t_z,  p_cb,  nullptr, nullptr, t_S,  64, 512, 1, 0, 1);
  cb_norms<<<2, 256, 0, stream>>>(cb, t_cn);
  vq_argmin<<<256, 256, 0, stream>>>(t_S, t_cn, t_ix);
  hipMemsetAsync(t_ls, 0, sizeof(float), stream);
  vq_gather_loss<<<256, 256, 0, stream>>>(t_ix, cb, t_z, t_qz, t_ls);

  // ---- decoder ----
  CONV( 9, 3, 1, 0, 6, 6, 6, 6, t_qz, p_dc1, dc1_b, nullptr,  t_p,  64, 128, 1, 1, 0);
  CONV( 9, 3, 1, 1, 6, 6, 6, 6, t_p,  p_dr1a, nullptr, nullptr, t_r, 128, 32, 1, 1, 0);
  CONV( 1, 1, 1, 1, 6, 6, 6, 6, t_r,  p_dr1b, nullptr, t_p,     t_q,  32, 128, 1, 0, 0);
  CONV( 9, 3, 1, 1, 6, 6, 6, 6, t_q,  p_dr2a, nullptr, nullptr, t_r, 128, 32, 1, 1, 0);
  CONV( 1, 1, 1, 1, 6, 6, 6, 6, t_r,  p_dr2b, nullptr, t_q,     t_p,  32, 128, 1, 0, 0);
  // dec_t1: transposed conv 128->64, out 128x128, via lhs_dilation=2, pad=2
  CONV(16, 4, 2, 0, 6, 6, 7, 7, t_p,  p_dt1, dt1_b, nullptr, t_a1, 128, 64, 1, 2, 0);
  dec_t2_direct<<<4096, 256, 0, stream>>>(t_a1, dt2_w, dt2_b, y_out + 1);

  // ---- loss scalar ----
  vq_finalize<<<1, 1, 0, stream>>>(t_ls, y_out);
}
